// PositionBlock_37684043055289
// MI455X (gfx1250) — matrix operand order repaired and checked by reading, not compiled, not run on hardware
//
#include <hip/hip_runtime.h>
#include <hip/hip_bf16.h>
#include <cstdint>
#include <cstddef>

#define CC    64
#define HH    80
#define WWID  120
#define BB    2
#define VV    3
#define BVN   6
#define DD    20
#define HW    (HH*WWID)
#define NEARP 0.1f
#define FARP  100.0f

typedef __attribute__((ext_vector_type(16))) __bf16 v16bf;
typedef __attribute__((ext_vector_type(8)))  __bf16 v8bf;
typedef __attribute__((ext_vector_type(8)))  float  v8f;
typedef __attribute__((ext_vector_type(4)))  unsigned int v4u32;
typedef __attribute__((ext_vector_type(8)))  int v8i32;
typedef __attribute__((ext_vector_type(4)))  int v4i32;

#if __has_builtin(__builtin_amdgcn_tensor_load_to_lds) && __has_builtin(__builtin_amdgcn_s_wait_tensorcnt)
#define USE_TDM 1
#else
#define USE_TDM 0
#endif

// ---- parameter block layout (float offsets inside workspace) ----
#define OFF_S1  0      // bn1 scale [67]
#define OFF_H1  67     // bn1 shift [67]
#define OFF_S2  134    // bn2 scale [64]
#define OFF_H2  198    // bn2 shift [64]
#define OFF_SCF 262    // bn_cf scale [128]
#define OFF_HCF 390    // bn_cf shift [128]
#define OFF_AX  518    // R_c2w @ Kinv per bv [6*9]
#define OFF_TX  572    // t_c2w per bv [6*3]
#define OFF_MW  590    // warp matrices [2*6*9]
#define OFF_TW  698    // warp translations [2*6*3]
#define P_FLOATS 768

__device__ __forceinline__ float bf16r(float x) { __bf16 b = (__bf16)x; return (float)b; }
__device__ __forceinline__ float silu(float x)  { return x / (1.f + __expf(-x)); }

__device__ void inv3(const float* a, float* o) {
  float d = a[0]*(a[4]*a[8]-a[5]*a[7]) - a[1]*(a[3]*a[8]-a[5]*a[6]) + a[2]*(a[3]*a[7]-a[4]*a[6]);
  float id = 1.f / d;
  o[0] =  (a[4]*a[8]-a[5]*a[7])*id;  o[1] = -(a[1]*a[8]-a[2]*a[7])*id;  o[2] =  (a[1]*a[5]-a[2]*a[4])*id;
  o[3] = -(a[3]*a[8]-a[5]*a[6])*id;  o[4] =  (a[0]*a[8]-a[2]*a[6])*id;  o[5] = -(a[0]*a[5]-a[2]*a[3])*id;
  o[6] =  (a[3]*a[7]-a[4]*a[6])*id;  o[7] = -(a[0]*a[7]-a[1]*a[6])*id;  o[8] =  (a[0]*a[4]-a[1]*a[3])*id;
}
__device__ void mm3(const float* a, const float* b, float* o) {
  for (int r = 0; r < 3; ++r)
    for (int c = 0; c < 3; ++c) {
      float s = 0.f;
      for (int k = 0; k < 3; ++k) s += a[r*3+k]*b[k*3+c];
      o[r*3+c] = s;
    }
}

#if USE_TDM
// Issue a 1 KB contiguous Tensor-DMA (TENSORcnt-tracked) global -> LDS copy.
// D# built per CDNA5 ISA ch.8: group0 {count=1, lds_addr, global_addr, type=2},
// group1 {data_size=2B, 1-D tile of 512 elems, tensor_dim0=512}, groups 2/3 unused.
__device__ __forceinline__ void tdm_load_1kb(const __bf16* gsrc, __bf16* ldst)
{
  unsigned long long ga = (unsigned long long)(uintptr_t)gsrc;
  unsigned int la = (unsigned int)(uintptr_t)ldst;   // LDS offset = low 32 bits of generic ptr
  v4u32 g0 = { 1u,                                     // count=1 (valid user D#)
               la,                                     // lds_addr (bytes)
               (unsigned int)ga,                       // global_addr[31:0]
               (unsigned int)(ga >> 32) | (2u << 30) };// global_addr[56:32] | type=2
  v8i32 g1 = { (int)0x00010000,   // data_size=1 -> 2-byte elements
               (int)0x02000000,   // tensor_dim0[15:0]=512 (bits 63:48)
               (int)0x00010000,   // tensor_dim1=1 (bits 111:80 low half)
               (int)0x02000000,   // tile_dim0=512 (bits 127:112)
               0,                 // tile_dim1=0 (1-D), tile_dim2=0
               512, 0, 0 };       // tensor_dim0_stride=512
  v4i32 gz = {0, 0, 0, 0};
#if defined(__clang_major__) && (__clang_major__ >= 23)
  v8i32 gz8 = {0, 0, 0, 0, 0, 0, 0, 0};
  __builtin_amdgcn_tensor_load_to_lds(g0, g1, gz, gz, gz8, 0);
#else
  __builtin_amdgcn_tensor_load_to_lds(g0, g1, gz, gz, 0);
#endif
}
#endif

// ---------------- setup: BN folds + all geometry matrices ----------------
__global__ void setup_kernel(const float* intr, const float* c2w,
                             const float* g1, const float* b1, const float* m1, const float* v1,
                             const float* g2, const float* b2, const float* m2, const float* v2,
                             const float* gcf, const float* bcf, const float* mcf, const float* vcf,
                             float* P)
{
  if (threadIdx.x != 0 || blockIdx.x != 0) return;
  for (int c = 0; c < 67; ++c)  { float s = g1[c]*rsqrtf(v1[c]+1e-5f);  P[OFF_S1+c]=s;  P[OFF_H1+c]=b1[c]-m1[c]*s; }
  for (int c = 0; c < 64; ++c)  { float s = g2[c]*rsqrtf(v2[c]+1e-5f);  P[OFF_S2+c]=s;  P[OFF_H2+c]=b2[c]-m2[c]*s; }
  for (int c = 0; c < 128; ++c) { float s = gcf[c]*rsqrtf(vcf[c]+1e-5f); P[OFF_SCF+c]=s; P[OFF_HCF+c]=bcf[c]-mcf[c]*s; }

  // world-xyz path: A = R_c2w(bf16) @ Kinv(bf16), t = t_c2w(bf16); b-major bv
  for (int i = 0; i < BVN; ++i) {
    float Ki[9]; inv3(intr + 9*i, Ki);
    for (int j = 0; j < 9; ++j) Ki[j] = bf16r(Ki[j]);
    const float* T = c2w + 16*i;
    float R[9], A[9];
    for (int r = 0; r < 3; ++r) for (int c = 0; c < 3; ++c) R[r*3+c] = bf16r(T[r*4+c]);
    mm3(R, Ki, A);
    for (int j = 0; j < 9; ++j) P[OFF_AX + 9*i + j] = A[j];
    for (int r = 0; r < 3; ++r) P[OFF_TX + 3*i + r] = bf16r(T[r*4+3]);
  }

  // plane-sweep warp matrices: M = Kc @ poseR @ Kc^-1, t = Kc @ pose_t ; (v,b)-major bv'
  for (int v = 0; v < VV; ++v)
    for (int b = 0; b < BB; ++b) {
      int bvp = v*BB + b;
      int src = b*VV + v;
      float Kc[9];
      const float srow[3] = {(float)WWID, (float)HH, 1.f};
      for (int r = 0; r < 3; ++r) for (int c = 0; c < 3; ++c) Kc[r*3+c] = intr[9*src + r*3+c]*srow[r];
      float Kci[9]; inv3(Kc, Kci);
      float Kcb[9], Kcib[9];
      for (int j = 0; j < 9; ++j) { Kcb[j] = bf16r(Kc[j]); Kcib[j] = bf16r(Kci[j]); }
      // rigid inverse of E[b,v]
      const float* Ev = c2w + 16*src;
      float Einv[16];
      for (int r = 0; r < 3; ++r) for (int c = 0; c < 3; ++c) Einv[r*4+c] = Ev[c*4+r];
      for (int r = 0; r < 3; ++r) { float s = 0.f; for (int c = 0; c < 3; ++c) s += Ev[c*4+r]*Ev[c*4+3]; Einv[r*4+3] = -s; }
      Einv[12]=Einv[13]=Einv[14]=0.f; Einv[15]=1.f;
      for (int sh = 0; sh < 2; ++sh) {
        int o = (v + sh + 1) % VV;
        const float* Eo = c2w + 16*(b*VV + o);
        float Pz[16];
        for (int r = 0; r < 4; ++r)
          for (int c = 0; c < 4; ++c) {
            float s = 0.f;
            for (int k = 0; k < 4; ++k) s += Eo[r*4+k]*Einv[k*4+c];
            Pz[r*4+c] = s;
          }
        float Rp[9], tp[3];
        for (int r = 0; r < 3; ++r) { for (int c = 0; c < 3; ++c) Rp[r*3+c] = bf16r(Pz[r*4+c]); tp[r] = bf16r(Pz[r*4+3]); }
        float tmp[9], M[9];
        mm3(Rp, Kcib, tmp); mm3(Kcb, tmp, M);
        for (int j = 0; j < 9; ++j) P[OFF_MW + (sh*BVN + bvp)*9 + j] = M[j];
        for (int r = 0; r < 3; ++r) {
          float s = 0.f;
          for (int c = 0; c < 3; ++c) s += Kcb[r*3+c]*tp[c];
          P[OFF_TW + (sh*BVN + bvp)*3 + r] = s;
        }
      }
    }
}

// ---------------- stage 1: depth -> world xyz, concat, BN1 fold -> bf16 ----------------
__launch_bounds__(256)
__global__ void stage1_kernel(const float* __restrict__ x, const float* __restrict__ P,
                              __bf16* __restrict__ hb)
{
  int idx = blockIdx.x*256 + threadIdx.x;
  if (idx >= BVN*HW) return;
  const int p  = idx % HW;
  const int bv = idx / HW;
  const int px = p % WWID, py = p / WWID;
  float x0v   = x[((size_t)bv*CC)*HW + p];
  float depth = 255.f*(0.5f*tanhf(x0v) + 0.5f);
  float db = bf16r(depth);
  float cu = bf16r(bf16r((float)px)*db);
  float cv = bf16r(bf16r((float)py)*db);
  const float* A = P + OFF_AX + bv*9;
  const float* t = P + OFF_TX + bv*3;
  float xyz[3];
  xyz[0] = A[0]*cu + A[1]*cv + A[2]*db + t[0];
  xyz[1] = A[3]*cu + A[4]*cv + A[5]*db + t[1];
  xyz[2] = A[6]*cu + A[7]*cv + A[8]*db + t[2];
  for (int c = 0; c < 3; ++c)
    hb[((size_t)bv*67 + c)*HW + p] = (__bf16)(xyz[c]*P[OFF_S1+c] + P[OFF_H1+c]);
  for (int c = 0; c < CC; ++c) {
    float val = x[((size_t)bv*CC + c)*HW + p];
    hb[((size_t)bv*67 + 3 + c)*HW + p] = (__bf16)(val*P[OFF_S1+3+c] + P[OFF_H1+3+c]);
  }
}

// ---------------- weight pre-pack into WMMA A-fragment layout ----------------
// A 16x32 bf16 layout: lane L: m=L&15, half=L>>4; elem e: vg=e>>1,pos=e&1;
//   k = half*16 + 2*vg + pos for all vg (contiguous: k = 16*half + e)
template<int CIN, int KS>
__launch_bounds__(256)
__global__ void pack_w(const float* __restrict__ w, __bf16* __restrict__ ap, int total)
{
  int idx = blockIdx.x*256 + threadIdx.x;
  if (idx >= total) return;
  int e     = idx & 15;
  int lane  = (idx >> 4) & 31;
  int frag  = idx >> 9;
  int mtile = frag & 3;
  int rsc   = frag >> 2;
  int rs     = rsc % (KS*KS);
  int cchunk = rsc / (KS*KS);
  int m    = mtile*16 + (lane & 15);
  int half = lane >> 4;
  int vg = e >> 1, pos = e & 1;
  int kloc = half*16 + 2*vg + pos;
  int c = cchunk*32 + kloc;
  int r = rs / KS, s = rs % KS;
  float val = (c < CIN) ? w[(((size_t)m*CIN + c)*KS + r)*KS + s] : 0.f;
  ap[idx] = (__bf16)val;
}

// ---------------- implicit-GEMM conv with v_wmma_f32_16x16x32_bf16 ----------------
// block: 256 thr = 8 waves = 4 Mtiles(64 out-ch) x 2 Ntiles(32 px in a row)
// Weights stream in via per-wave double-buffered TDM (TENSORcnt), inputs staged
// channel-last in LDS so B-fragments are two ds_load_b128 per lane.
// EPI 0: silu+BN2 -> bf16    EPI 1: silu -> bf16 ch-major + ch-last
// EPI 2: +bias, BNcf, vb->b remap -> fused bf16    EPI 3: silu -> f32 out
template<int CIN, int KS, int EPI>
__launch_bounds__(256)
__global__ void conv_wmma(const __bf16* __restrict__ in,
                          const __bf16* __restrict__ Apack,
                          const float*  __restrict__ bias,
                          const float*  __restrict__ P,
                          __bf16* __restrict__ outb,
                          __bf16* __restrict__ outb2,
                          float*  __restrict__ outf)
{
  constexpr int PAD    = KS/2;
  constexpr int LDSW   = 32 + KS - 1;
  constexpr int NCHUNK = (CIN + 31)/32;
  constexpr int QUADS  = (LDSW + 3)/4;
  __shared__ alignas(32) __bf16 lds[KS*LDSW*32];   // channel-last [row][col][32 cin]
#if USE_TDM
  __shared__ alignas(32) __bf16 abuf[8][2][512];   // per-wave A-fragment double buffer (2x1KB)
#endif

  const int tid = threadIdx.x;
  const int NTB = (WWID + 31)/32;
  const int tb  = blockIdx.x % NTB;
  const int h0  = (blockIdx.x / NTB) % HH;
  const int bv  = blockIdx.x / (NTB*HH);
  const int w0  = tb*32;
  const int lane  = tid & 31;
  const int wid   = tid >> 5;
  const int mtile = wid & 3;
  const int ntile = wid >> 2;
  const int n     = lane & 15;
  const int khalf = lane >> 4;

  v8f acc = {0.f,0.f,0.f,0.f,0.f,0.f,0.f,0.f};

  for (int cchunk = 0; cchunk < NCHUNK; ++cchunk) {
    __syncthreads();
    const int c0 = cchunk*32;
    for (int idx = tid; idx < 32*KS*QUADS; idx += 256) {
      int cl   = idx / (KS*QUADS);
      int rem  = idx % (KS*QUADS);
      int row  = rem / QUADS;
      int col0 = (rem % QUADS)*4;
      int c  = c0 + cl;
      int hh = h0 - PAD + row;
      __bf16 vals[4];
      vals[0]=vals[1]=vals[2]=vals[3]=(__bf16)0.f;
      if (c < CIN && hh >= 0 && hh < HH) {
        const __bf16* src = in + (((size_t)bv*CIN + c)*HH + hh)*WWID;
        int wbase = w0 - PAD + col0;
        if (((wbase & 3) == 0) && wbase >= 0 && wbase + 3 < WWID) {
          union { unsigned long long u; __bf16 h[4]; } uu;
          uu.u = *(const unsigned long long*)(src + wbase);
          vals[0]=uu.h[0]; vals[1]=uu.h[1]; vals[2]=uu.h[2]; vals[3]=uu.h[3];
        } else {
          for (int j = 0; j < 4; ++j) { int ww = wbase + j; if (ww >= 0 && ww < WWID) vals[j] = src[ww]; }
        }
      }
      for (int j = 0; j < 4; ++j) { int col = col0 + j; if (col < LDSW) lds[(row*LDSW + col)*32 + cl] = vals[j]; }
    }
    if (cchunk + 1 < NCHUNK)  // pull next chunk's packed weights toward L2 (global_prefetch)
      __builtin_prefetch(Apack + ((size_t)((cchunk+1)*KS*KS)*4)*512 + (size_t)tid*64, 0, 1);
    __syncthreads();

#if USE_TDM
    // kick off the chunk's first A-fragment DMA (wave-private, TENSORcnt-tracked)
    tdm_load_1kb(Apack + (((size_t)(cchunk*KS*KS + 0)*4 + mtile) << 9), &abuf[wid][0][0]);
#endif
#pragma unroll 9
    for (int rs = 0; rs < KS*KS; ++rs) {
      const int r = rs / KS, s = rs % KS;
#if USE_TDM
      if (rs + 1 < KS*KS) {
        tdm_load_1kb(Apack + (((size_t)(cchunk*KS*KS + rs + 1)*4 + mtile) << 9),
                     &abuf[wid][(rs + 1) & 1][0]);
        __builtin_amdgcn_s_wait_tensorcnt(1);   // fragment rs resident, rs+1 in flight
      } else {
        __builtin_amdgcn_s_wait_tensorcnt(0);   // drain
      }
      v16bf afrag = *(const v16bf*)(&abuf[wid][rs & 1][lane*16]);
#else
      const __bf16* ap = Apack + (((size_t)(cchunk*KS*KS + rs)*4 + mtile)*32 + lane)*16;
      v16bf afrag = *(const v16bf*)ap;
#endif
      const __bf16* bp = lds + ((r*LDSW + (ntile*16 + n + s))*32 + khalf*16);
      v16bf bfrag = *(const v16bf*)bp;
      acc = __builtin_amdgcn_wmma_f32_16x16x32_bf16(false, afrag, false, bfrag, (short)0, acc, false, false);
    }
  }

  const int w = w0 + ntile*16 + n;
  if (w >= WWID) return;
  const size_t ppix = (size_t)h0*WWID + w;
#pragma unroll
  for (int r = 0; r < 8; ++r) {
    const int m = mtile*16 + khalf*8 + r;   // C/D layout: VGPR r -> M = r + 8*(lane>=16)
    float val = acc[r] + bias[m];
    if (EPI == 0) {
      val = silu(val);
      val = val*P[OFF_S2+m] + P[OFF_H2+m];
      outb[((size_t)bv*64 + m)*HW + ppix] = (__bf16)val;
    } else if (EPI == 1) {
      val = silu(val);
      __bf16 hv = (__bf16)val;
      outb [((size_t)bv*64 + m)*HW + ppix] = hv;          // channel-major
      outb2[((size_t)bv*HW + ppix)*64 + m] = hv;          // channel-last (for sampling)
    } else if (EPI == 2) {
      int vv = bv / BB, b = bv % BB;                      // here bv is (v,b)-major
      int bvo = b*VV + vv;
      val = val*P[OFF_SCF+m] + P[OFF_HCF+m];
      outb[((size_t)bvo*128 + m)*HW + ppix] = (__bf16)val;
    } else {
      val = silu(val);
      outf[((size_t)bv*64 + m)*HW + ppix] = val;
    }
  }
}

// ---------------- plane-sweep correlation (bilinear gather + 64-ch dot) ----------------
__launch_bounds__(256)
__global__ void corr_kernel(const __bf16* __restrict__ hcl, const float* __restrict__ P,
                            __bf16* __restrict__ cin)
{
  int idx = blockIdx.x*256 + threadIdx.x;
  if (idx >= BVN*DD*HW) return;
  const int p   = idx % HW;
  const int d   = (idx / HW) % DD;
  const int bvp = idx / (HW*DD);          // (v,b)-major
  const int v = bvp / BB, b = bvp % BB;
  const int px = p % WWID, py = p / WWID;
  const float dmin = 1.f/FARP, dmax = 1.f/NEARP;
  const float depth = 1.f/(dmin + ((float)d/(DD-1))*(dmax - dmin));

  const __bf16* f01 = hcl + ((size_t)(b*VV+v)*HW + p)*64;
  v8bf f[8];
#pragma unroll
  for (int i = 0; i < 8; ++i) f[i] = *(const v8bf*)(f01 + i*8);

  float total = 0.f;
#pragma unroll
  for (int sh = 0; sh < 2; ++sh) {
    const float* M = P + OFF_MW + (sh*BVN + bvp)*9;
    const float* T = P + OFF_TW + (sh*BVN + bvp)*3;
    float X = (M[0]*px + M[1]*py + M[2])*depth + T[0];
    float Y = (M[3]*px + M[4]*py + M[5])*depth + T[1];
    float Z = (M[6]*px + M[7]*py + M[8])*depth + T[2];
    float den = fmaxf(Z, 1e-3f);
    float fx = X/den, fy = Y/den;               // == (grid+1)/2*(dim-1)
    float x0f = floorf(fx), y0f = floorf(fy);
    int x0 = (int)x0f, y0 = (int)y0f;
    float wx = fx - x0f, wy = fy - y0f;
    const __bf16* base = hcl + (size_t)(b*VV + ((v + sh + 1) % VV))*HW*64;
    float dsum = 0.f;
#pragma unroll
    for (int t = 0; t < 4; ++t) {
      int xi = x0 + (t & 1), yi = y0 + (t >> 1);
      float wgt = ((t & 1) ? wx : 1.f - wx) * ((t >> 1) ? wy : 1.f - wy);
      if (xi >= 0 && xi < WWID && yi >= 0 && yi < HH) {
        const __bf16* g = base + ((size_t)yi*WWID + xi)*64;
        float s = 0.f;
#pragma unroll
        for (int i = 0; i < 8; ++i) {
          v8bf tv = *(const v8bf*)(g + i*8);
#pragma unroll
          for (int j = 0; j < 8; ++j) s += (float)f[i][j]*(float)tv[j];
        }
        dsum += wgt*s;
      }
    }
    total += dsum;
  }
  cin[((size_t)bvp*84 + d)*HW + p] = (__bf16)(total * (1.f/16.f));  // / sqrt(64) / (V-1)
}

// ---------------- packing helpers ----------------
__launch_bounds__(256)
__global__ void cinpack_kernel(const __bf16* __restrict__ hcm, __bf16* __restrict__ cin)
{
  int idx = blockIdx.x*256 + threadIdx.x;
  if (idx >= BVN*CC*HW) return;
  int p   = idx % HW;
  int c   = (idx / HW) % CC;
  int bvp = idx / (HW*CC);          // (v,b)-major
  int v = bvp / BB, b = bvp % BB;
  cin[((size_t)bvp*84 + 20 + c)*HW + p] = hcm[((size_t)(b*VV+v)*CC + c)*HW + p];
}

__launch_bounds__(256)
__global__ void fusedpack_kernel(const __bf16* __restrict__ hcm, const float* __restrict__ P,
                                 __bf16* __restrict__ fused)
{
  int idx = blockIdx.x*256 + threadIdx.x;
  if (idx >= BVN*CC*HW) return;
  int p  = idx % HW;
  int c  = (idx / HW) % CC;
  int bv = idx / (HW*CC);           // b-major
  int ch = 64 + c;
  float val = (float)hcm[((size_t)bv*CC + c)*HW + p];
  fused[((size_t)bv*128 + ch)*HW + p] = (__bf16)(val*P[OFF_SCF+ch] + P[OFF_HCF+ch]);
}

// ---------------- host side ----------------
extern "C" void kernel_launch(void* const* d_in, const int* in_sizes, int n_in,
                              void* d_out, int out_size, void* d_ws, size_t ws_size,
                              hipStream_t stream)
{
  (void)in_sizes; (void)n_in; (void)out_size; (void)ws_size;
  const float* x    = (const float*)d_in[0];
  const float* intr = (const float*)d_in[1];
  const float* c2w  = (const float*)d_in[2];
  const float* g1   = (const float*)d_in[3];
  const float* b1   = (const float*)d_in[4];
  const float* m1   = (const float*)d_in[5];
  const float* v1   = (const float*)d_in[6];
  const float* w1   = (const float*)d_in[7];
  const float* pb1  = (const float*)d_in[8];
  const float* g2   = (const float*)d_in[9];
  const float* b2   = (const float*)d_in[10];
  const float* m2   = (const float*)d_in[11];
  const float* v2   = (const float*)d_in[12];
  const float* w2   = (const float*)d_in[13];
  const float* pb2  = (const float*)d_in[14];
  const float* cpw  = (const float*)d_in[15];
  const float* cpb  = (const float*)d_in[16];
  const float* gcf  = (const float*)d_in[17];
  const float* bcf  = (const float*)d_in[18];
  const float* mcf  = (const float*)d_in[19];
  const float* vcf  = (const float*)d_in[20];
  const float* cfw  = (const float*)d_in[21];
  const float* cfb  = (const float*)d_in[22];
  float* out = (float*)d_out;

  uint8_t* wsb = (uint8_t*)d_ws;
  size_t off = 0;
  auto alloc = [&](size_t bytes) -> void* {
    off = (off + 255) & ~(size_t)255;
    void* p = wsb + off;
    off += bytes;
    return p;
  };
  float*  P     = (float*) alloc((size_t)P_FLOATS*4);
  __bf16* hb    = (__bf16*)alloc((size_t)BVN*67*HW*2);   // bn1(concat(xyz,x))
  __bf16* h1b   = (__bf16*)alloc((size_t)BVN*64*HW*2);   // bn2(silu(conv1))
  __bf16* hcm   = (__bf16*)alloc((size_t)BVN*64*HW*2);   // h channel-major
  __bf16* hcl   = (__bf16*)alloc((size_t)BVN*64*HW*2);   // h channel-last
  __bf16* cin   = (__bf16*)alloc((size_t)BVN*84*HW*2);   // [corr | feat01]
  __bf16* fused = (__bf16*)alloc((size_t)BVN*128*HW*2);  // bncf([cost | xr])
  __bf16* ap1   = (__bf16*)alloc((size_t)3*81*2048*2);
  __bf16* ap2   = (__bf16*)alloc((size_t)2*81*2048*2);
  __bf16* apcp  = (__bf16*)alloc((size_t)3*9 *2048*2);
  __bf16* apcf  = (__bf16*)alloc((size_t)4*81*2048*2);

  auto cdiv = [](int a, int b) { return (a + b - 1)/b; };

  setup_kernel<<<1, 1, 0, stream>>>(intr, c2w, g1,b1,m1,v1, g2,b2,m2,v2, gcf,bcf,mcf,vcf, P);

  { int tot = 3*81*2048; pack_w<67,9> <<<cdiv(tot,256),256,0,stream>>>(w1,  ap1,  tot); }
  { int tot = 2*81*2048; pack_w<64,9> <<<cdiv(tot,256),256,0,stream>>>(w2,  ap2,  tot); }
  { int tot = 3*9 *2048; pack_w<84,3> <<<cdiv(tot,256),256,0,stream>>>(cpw, apcp, tot); }
  { int tot = 4*81*2048; pack_w<128,9><<<cdiv(tot,256),256,0,stream>>>(cfw, apcf, tot); }

  stage1_kernel<<<cdiv(BVN*HW,256),256,0,stream>>>(x, P, hb);

  const int NTB = (WWID + 31)/32;
  const int convGrid = BVN*HH*NTB;
  conv_wmma<67,9,0><<<convGrid,256,0,stream>>>(hb,  ap1, pb1, P, h1b, nullptr, nullptr);
  conv_wmma<64,9,1><<<convGrid,256,0,stream>>>(h1b, ap2, pb2, P, hcm, hcl,    nullptr);

  corr_kernel   <<<cdiv(BVN*DD*HW,256),256,0,stream>>>(hcl, P, cin);
  cinpack_kernel<<<cdiv(BVN*CC*HW,256),256,0,stream>>>(hcm, cin);

  conv_wmma<84,3,2><<<convGrid,256,0,stream>>>(cin, apcp, cpb, P, fused, nullptr, nullptr);
  fusedpack_kernel<<<cdiv(BVN*CC*HW,256),256,0,stream>>>(hcm, P, fused);

  conv_wmma<128,9,3><<<convGrid,256,0,stream>>>(fused, apcf, cfb, P, nullptr, nullptr, out);
}
